// MambaBlock_49237505081427
// MI455X (gfx1250) — compile-verified, matrix operand order checked by reading, not run on hardware
//
#include <hip/hip_runtime.h>
#include <hip/hip_bf16.h>

// ---------------------------------------------------------------------------
// Mamba block for gfx1250 (MI455X). B=2, L=2048, DIM=1024, D_INNER=2048,
// DT_RANK=64, D_STATE=16, D_CONV=4.  GEMMs run on v_wmma_f32_16x16x32_f16
// with 2x4 register tiling (8 WMMA per 12 b128 loads per K-chunk).
// ---------------------------------------------------------------------------

typedef __attribute__((ext_vector_type(16))) _Float16 v16h;
typedef __attribute__((ext_vector_type(8)))  _Float16 v8h;
typedef __attribute__((ext_vector_type(8)))  float    v8f;

#define DIM_      1024
#define DSTATE_   16
#define DCONV_    4
#define DINNER_   2048
#define DTRANK_   64
#define BSZ_      2
#define LSEQ_     2048
#define ROWS_     (BSZ_ * LSEQ_)           // 4096
#define XPROJ_N_  (DTRANK_ + 2 * DSTATE_)  // 96

// ---------------------------------------------------------------------------
// Convert + transpose f32 [K,N] row-major -> f16 [N,K] row-major, so that a
// WMMA B-fragment (column of B) becomes a contiguous K-run, matching the
// A-fragment load pattern (two 16B loads per lane).
// ---------------------------------------------------------------------------
__global__ void cvt_transpose_f16(const float* __restrict__ W,
                                  _Float16* __restrict__ Wt,
                                  int K, int N) {
  int gid = blockIdx.x * blockDim.x + threadIdx.x;
  if (gid >= K * N) return;
  int k = gid / N;
  int n = gid - k * N;
  Wt[(size_t)n * K + k] = (_Float16)W[gid];
}

// ---------------------------------------------------------------------------
// LayerNorm over last dim (1024), output f16.
// ---------------------------------------------------------------------------
__global__ void layernorm_f16(const float* __restrict__ x,
                              const float* __restrict__ gamma,
                              const float* __restrict__ beta,
                              _Float16* __restrict__ xn) {
  __shared__ float red[64];
  const int row = blockIdx.x;
  const float* xr = x + (size_t)row * DIM_;
  float s = 0.f, ss = 0.f;
  for (int i = threadIdx.x; i < DIM_; i += blockDim.x) {
    float v = xr[i];
    s += v; ss += v * v;
  }
  for (int off = 16; off > 0; off >>= 1) {
    s  += __shfl_down(s,  off, 32);
    ss += __shfl_down(ss, off, 32);
  }
  const int wid = threadIdx.x >> 5;
  const int nw  = blockDim.x >> 5;
  if ((threadIdx.x & 31) == 0) { red[wid] = s; red[32 + wid] = ss; }
  __syncthreads();
  if (threadIdx.x == 0) {
    float ts = 0.f, tss = 0.f;
    for (int w = 0; w < nw; ++w) { ts += red[w]; tss += red[32 + w]; }
    red[0] = ts; red[32] = tss;
  }
  __syncthreads();
  const float mu  = red[0]  * (1.0f / DIM_);
  const float var = red[32] * (1.0f / DIM_) - mu * mu;
  const float inv = rsqrtf(var + 1e-5f);
  for (int i = threadIdx.x; i < DIM_; i += blockDim.x) {
    float v = (xr[i] - mu) * inv * gamma[i] + beta[i];
    xn[(size_t)row * DIM_ + i] = (_Float16)v;
  }
}

// ---------------------------------------------------------------------------
// WMMA GEMM: C[M,N] = A[M,K](f16 row-major) x Bt[N,K](f16, transposed weight).
// One wave -> BM x BN 16x16 tiles; K-loop step 32 on v_wmma_f32_16x16x32_f16.
// Fragment layout per CDNA5 ISA 7.12.2:
//   lane<16 : K = {k0..k0+7, k0+16..k0+23};  lane>=16: K = {k0+8..15, k0+24..31}
//   C/D: VGPR v <-> row = hi*8 + v, col = lane%16.
// MODE 0: store f16              (xz = in_proj)
// MODE 1: store f32              (x_dbl = x_proj)
// MODE 2: softplus(acc+bias)->f16 (delta = dt_proj)
// MODE 3: acc + residual -> f32   (out_proj + residual)
// ---------------------------------------------------------------------------
template <int MODE, int BM, int BN>
__global__ __launch_bounds__(128)
void gemm_wmma_f16(const _Float16* __restrict__ A,
                   const _Float16* __restrict__ Bt,
                   void* __restrict__ Cout,
                   const float* __restrict__ aux,   // bias (MODE2) / residual (MODE3)
                   int M, int N, int K) {
  const int lane = threadIdx.x;        // 0..31
  const int m    = lane & 15;
  const int hi   = lane >> 4;          // 0 or 1
  const int tileN0 = blockIdx.x * BN;
  const int tileM0 = (blockIdx.y * blockDim.y + threadIdx.y) * BM;

  const _Float16* __restrict__ arow[BM];
  const _Float16* __restrict__ brow[BN];
#pragma unroll
  for (int i = 0; i < BM; ++i)
    arow[i] = A + (size_t)((tileM0 + i) * 16 + m) * K;
#pragma unroll
  for (int j = 0; j < BN; ++j)
    brow[j] = Bt + (size_t)((tileN0 + j) * 16 + m) * K;

  v8f acc[BM][BN] = {};
  const int klo = hi * 8;

  for (int k0 = 0; k0 < K; k0 += 32) {
    v16h a[BM], b[BN];
#pragma unroll
    for (int i = 0; i < BM; ++i) {
      v8h lo  = *(const v8h*)(arow[i] + k0 + klo);
      v8h hi8 = *(const v8h*)(arow[i] + k0 + 16 + klo);
#pragma unroll
      for (int t = 0; t < 8; ++t) { a[i][t] = lo[t]; a[i][8 + t] = hi8[t]; }
    }
#pragma unroll
    for (int j = 0; j < BN; ++j) {
      v8h lo  = *(const v8h*)(brow[j] + k0 + klo);
      v8h hi8 = *(const v8h*)(brow[j] + k0 + 16 + klo);
#pragma unroll
      for (int t = 0; t < 8; ++t) { b[j][t] = lo[t]; b[j][8 + t] = hi8[t]; }
    }
    if (k0 + 32 < K) {   // pull next K-chunk toward the WGP (global_prefetch_b8)
#pragma unroll
      for (int i = 0; i < BM; ++i) __builtin_prefetch(arow[i] + k0 + 32 + klo, 0, 0);
#pragma unroll
      for (int j = 0; j < BN; ++j) __builtin_prefetch(brow[j] + k0 + 32 + klo, 0, 0);
    }
#pragma unroll
    for (int i = 0; i < BM; ++i)
#pragma unroll
      for (int j = 0; j < BN; ++j)
        acc[i][j] = __builtin_amdgcn_wmma_f32_16x16x32_f16(
            /*neg_a=*/false, a[i], /*neg_b=*/false, b[j],
            /*c_mod=*/(short)0, acc[i][j], /*reuse_a=*/false, /*reuse_b=*/false);
  }

#pragma unroll
  for (int i = 0; i < BM; ++i) {
#pragma unroll
    for (int j = 0; j < BN; ++j) {
      const int col = (tileN0 + j) * 16 + m;
#pragma unroll
      for (int v = 0; v < 8; ++v) {
        const int row = (tileM0 + i) * 16 + hi * 8 + v;
        const size_t idx = (size_t)row * N + col;
        const float r = acc[i][j][v];
        if (MODE == 0) {
          ((_Float16*)Cout)[idx] = (_Float16)r;
        } else if (MODE == 1) {
          ((float*)Cout)[idx] = r;
        } else if (MODE == 2) {
          const float xb = r + aux[col];
          const float sp = (xb > 20.f) ? xb : __logf(1.f + __expf(xb));
          ((_Float16*)Cout)[idx] = (_Float16)sp;
        } else {
          ((float*)Cout)[idx] = r + aux[idx];
        }
      }
    }
  }
}

// ---------------------------------------------------------------------------
// Depthwise causal conv1d (k=4) + bias + SiLU over the u-half of xz.
// xz: [ROWS, 2*DINNER] f16 (cols 0..2047 = u_pre, 2048..4095 = z).
// ---------------------------------------------------------------------------
__global__ void conv_silu(const _Float16* __restrict__ xz,
                          const float* __restrict__ cw,   // [DINNER,1,4]
                          const float* __restrict__ cb,   // [DINNER]
                          _Float16* __restrict__ u) {
  const int gid = blockIdx.x * blockDim.x + threadIdx.x;   // over ROWS*DINNER
  if (gid >= ROWS_ * DINNER_) return;
  const int d  = gid & (DINNER_ - 1);
  const int rl = gid >> 11;                 // b*L + l
  const int l  = rl & (LSEQ_ - 1);
  float acc = cb[d];
#pragma unroll
  for (int j = 0; j < DCONV_; ++j) {
    const int lj = l - (DCONV_ - 1) + j;
    if (lj >= 0)
      acc += cw[d * DCONV_ + j] *
             (float)xz[(size_t)(rl - (DCONV_ - 1) + j) * (2 * DINNER_) + d];
  }
  const float s = acc / (1.f + __expf(-acc));  // silu
  u[(size_t)rl * DINNER_ + d] = (_Float16)s;
}

// ---------------------------------------------------------------------------
// Extract dt (first 64 cols of x_dbl) as f16 row-major [ROWS, 64] for dt_proj.
// ---------------------------------------------------------------------------
__global__ void extract_dt(const float* __restrict__ xdbl,
                           _Float16* __restrict__ dt16) {
  const int gid = blockIdx.x * blockDim.x + threadIdx.x;   // ROWS*64
  if (gid >= ROWS_ * DTRANK_) return;
  const int r = gid >> 6;
  const int c = gid & (DTRANK_ - 1);
  dt16[gid] = (_Float16)xdbl[(size_t)r * XPROJ_N_ + c];
}

// ---------------------------------------------------------------------------
// Selective scan over time, fused with D-skip, z-gate (SiLU) and f16 output.
// One thread per (b, d) channel; h[16], A[16] live in VGPRs; serial in t.
// This is the true critical path (L=2048 dependent steps, 128 waves of
// parallelism); B/C rows (128B per t) stay hot in L2.
// ---------------------------------------------------------------------------
__global__ __launch_bounds__(256)
void selective_scan(const _Float16* __restrict__ u,
                    const _Float16* __restrict__ delta,
                    const float* __restrict__ xdbl,    // [ROWS,96]: dt|B|C
                    const _Float16* __restrict__ xz,   // z in cols 2048..4095
                    const float* __restrict__ A_log,   // [DINNER,16]
                    const float* __restrict__ Dp,      // [DINNER]
                    _Float16* __restrict__ y) {
  const int gid = blockIdx.x * blockDim.x + threadIdx.x;  // 0..B*DINNER-1
  const int b = gid >> 11;
  const int d = gid & (DINNER_ - 1);
  float A[DSTATE_], h[DSTATE_];
#pragma unroll
  for (int s = 0; s < DSTATE_; ++s) {
    A[s] = -__expf(A_log[d * DSTATE_ + s]);
    h[s] = 0.f;
  }
  const float Dv = Dp[d];
  for (int t = 0; t < LSEQ_; ++t) {
    const size_t r = (size_t)b * LSEQ_ + t;
    const float ut = (float)u[r * DINNER_ + d];
    const float dt = (float)delta[r * DINNER_ + d];
    const float* __restrict__ bc = xdbl + r * XPROJ_N_;
    float yt = 0.f;
#pragma unroll
    for (int s = 0; s < DSTATE_; ++s) {
      h[s] = h[s] * __expf(dt * A[s]) + dt * bc[DTRANK_ + s] * ut;
      yt  += h[s] * bc[DTRANK_ + DSTATE_ + s];
    }
    yt += ut * Dv;
    const float z = (float)xz[r * (2 * DINNER_) + DINNER_ + d];
    yt *= z / (1.f + __expf(-z));
    y[r * DINNER_ + d] = (_Float16)yt;
  }
}

// ---------------------------------------------------------------------------
// Host side
// ---------------------------------------------------------------------------
extern "C" void kernel_launch(void* const* d_in, const int* in_sizes, int n_in,
                              void* d_out, int out_size, void* d_ws, size_t ws_size,
                              hipStream_t stream) {
  (void)in_sizes; (void)n_in; (void)out_size; (void)ws_size;
  const float* x      = (const float*)d_in[0];   // [B,L,1024]
  const float* gamma  = (const float*)d_in[1];
  const float* beta   = (const float*)d_in[2];
  const float* W_in   = (const float*)d_in[3];   // [1024, 4096]
  const float* conv_w = (const float*)d_in[4];   // [2048,1,4]
  const float* conv_b = (const float*)d_in[5];
  const float* W_xp   = (const float*)d_in[6];   // [2048, 96]
  const float* W_dt   = (const float*)d_in[7];   // [64, 2048]
  const float* b_dt   = (const float*)d_in[8];
  const float* A_log  = (const float*)d_in[9];   // [2048,16]
  const float* Dp     = (const float*)d_in[10];
  const float* W_out  = (const float*)d_in[11];  // [2048, 1024]
  float* out = (float*)d_out;

  // Carve workspace (aligned to 256B).
  uint8_t* ws = (uint8_t*)d_ws;
  size_t off = 0;
  auto carve = [&](size_t bytes) -> uint8_t* {
    uint8_t* p = ws + off;
    off += (bytes + 255) & ~(size_t)255;
    return p;
  };
  _Float16* Win_t  = (_Float16*)carve((size_t)DIM_ * 2 * DINNER_ * 2);   // [4096,1024]
  _Float16* Wxp_t  = (_Float16*)carve((size_t)XPROJ_N_ * DINNER_ * 2);   // [96,2048]
  _Float16* Wdt_t  = (_Float16*)carve((size_t)DINNER_ * DTRANK_ * 2);    // [2048,64]
  _Float16* Wout_t = (_Float16*)carve((size_t)DIM_ * DINNER_ * 2);       // [1024,2048]
  _Float16* xn     = (_Float16*)carve((size_t)ROWS_ * DIM_ * 2);
  _Float16* xzbuf  = (_Float16*)carve((size_t)ROWS_ * 2 * DINNER_ * 2);
  _Float16* ubuf   = (_Float16*)carve((size_t)ROWS_ * DINNER_ * 2);
  float*    xdbl   = (float*)   carve((size_t)ROWS_ * XPROJ_N_ * 4);
  _Float16* dt16   = (_Float16*)carve((size_t)ROWS_ * DTRANK_ * 2);
  _Float16* delta  = (_Float16*)carve((size_t)ROWS_ * DINNER_ * 2);
  _Float16* ybuf   = (_Float16*)carve((size_t)ROWS_ * DINNER_ * 2);

  const dim3 gblk(32, 4);  // 4 waves per block

  // 0) weight convert+transpose
  {
    int n;
    n = DIM_ * 2 * DINNER_;
    cvt_transpose_f16<<<(n + 255) / 256, 256, 0, stream>>>(W_in, Win_t, DIM_, 2 * DINNER_);
    n = DINNER_ * XPROJ_N_;
    cvt_transpose_f16<<<(n + 255) / 256, 256, 0, stream>>>(W_xp, Wxp_t, DINNER_, XPROJ_N_);
    n = DTRANK_ * DINNER_;
    cvt_transpose_f16<<<(n + 255) / 256, 256, 0, stream>>>(W_dt, Wdt_t, DTRANK_, DINNER_);
    n = DINNER_ * DIM_;
    cvt_transpose_f16<<<(n + 255) / 256, 256, 0, stream>>>(W_out, Wout_t, DINNER_, DIM_);
  }

  // 1) LayerNorm -> xn (f16)
  layernorm_f16<<<ROWS_, 256, 0, stream>>>(x, gamma, beta, xn);

  // 2) in_proj: xz = xn x W_in   [4096,1024]x[1024,4096] -> f16
  //    2x4 tiles/wave: grid (4096/(16*4), 4096/(16*2*4)) = (64, 32)
  gemm_wmma_f16<0, 2, 4><<<dim3((2 * DINNER_) / 64, ROWS_ / 128), gblk, 0, stream>>>(
      xn, Win_t, xzbuf, nullptr, ROWS_, 2 * DINNER_, DIM_);

  // 3) depthwise causal conv + SiLU -> u (f16)
  conv_silu<<<(ROWS_ * DINNER_) / 256, 256, 0, stream>>>(xzbuf, conv_w, conv_b, ubuf);

  // 4) x_proj: x_dbl = u x W_xproj   [4096,2048]x[2048,96] -> f32
  //    N = 96 = 6 tiles -> BN=2, grid x = 3
  gemm_wmma_f16<1, 2, 2><<<dim3(XPROJ_N_ / 32, ROWS_ / 128), gblk, 0, stream>>>(
      ubuf, Wxp_t, xdbl, nullptr, ROWS_, XPROJ_N_, DINNER_);

  // 5) dt slice -> f16
  extract_dt<<<(ROWS_ * DTRANK_) / 256, 256, 0, stream>>>(xdbl, dt16);

  // 6) dt_proj + softplus: delta = softplus(dt x W_dt + b_dt) -> f16
  gemm_wmma_f16<2, 2, 4><<<dim3(DINNER_ / 64, ROWS_ / 128), gblk, 0, stream>>>(
      dt16, Wdt_t, delta, b_dt, ROWS_, DINNER_, DTRANK_);

  // 7) selective scan + D-skip + z-gate -> y (f16)
  selective_scan<<<(BSZ_ * DINNER_) / 256, 256, 0, stream>>>(
      ubuf, delta, xdbl, xzbuf, A_log, Dp, ybuf);

  // 8) out_proj + residual: out = y x W_out + x  -> f32
  gemm_wmma_f16<3, 2, 4><<<dim3(DIM_ / 64, ROWS_ / 128), gblk, 0, stream>>>(
      ybuf, Wout_t, out, x, ROWS_, DIM_, DINNER_);
}
